// Phase3GATModel_50113678410323
// MI455X (gfx1250) — matrix operand order repaired and checked by reading, not compiled, not run on hardware
//
#include <hip/hip_runtime.h>

// ---- problem constants (match reference) ----
#define B    8
#define Nn   4096
#define Ee   65536
#define Hh   4
#define DHd  16
#define GH   64
#define EEMB 16
#define DF   1384

typedef __attribute__((ext_vector_type(16))) _Float16 v16h;
typedef __attribute__((ext_vector_type(8)))  float    v8f;

__device__ __forceinline__ float gelu_exact(float x) {
    return 0.5f * x * (1.0f + erff(x * 0.70710678118654752440f));
}

// sign-aware float atomic max via int/uint compare (monotonic float<->int mapping)
__device__ __forceinline__ void atomicMaxF(float* addr, float v) {
    if (v >= 0.0f) atomicMax((int*)addr, __float_as_int(v));
    else           atomicMin((unsigned int*)addr, __float_as_uint(v));
}

// ---------------- node feature projection: x = nf @ Wᵀ + b  [B*N, GH] ----------------
__global__ void k_node_proj(const float* __restrict__ nf, const float* __restrict__ w,
                            const float* __restrict__ bias, float* __restrict__ x) {
    int idx = blockIdx.x * blockDim.x + threadIdx.x;          // over B*Nn*GH
    int gh   = idx & (GH - 1);
    int node = idx >> 6;                                      // b*Nn + n
    const float* f  = nf + (size_t)node * 22;
    const float* wr = w + gh * 22;
    float s = bias[gh];
#pragma unroll
    for (int k = 0; k < 22; ++k) s += f[k] * wr[k];
    x[idx] = s;
}

// ---------------- h = x @ Wₗᵀ via v_wmma_f32_16x16x32_f16 ----------------
// One wave per 16-row strip; A (16x64) loaded once into registers and reused across
// all 4 column tiles (8 WMMAs per wave). W converted to f16 once per block into LDS;
// B operands read as 32B-aligned ds_load_b128.
__global__ void k_gemm_h(const float* __restrict__ x, const float* __restrict__ gatW,
                         int layer, float* __restrict__ h) {
    __shared__ __align__(16) _Float16 Wlds[GH * GH];          // 8 KB of the 320 KB WGP LDS
    const float* W = gatW + (size_t)layer * GH * GH;          // [out][in] row-major
    // cooperative f32 -> f16 conversion of W (once per block)
    for (int j = threadIdx.x; j < GH * GH; j += 256) Wlds[j] = (_Float16)W[j];
    __syncthreads();

    int strip = blockIdx.x * (blockDim.x >> 5) + (threadIdx.x >> 5);  // b*256 + mt
    int lane  = threadIdx.x & 31;
    int mt = strip & 255;          // Nn/16 row strips
    int b  = strip >> 8;
    int m  = lane & 15;            // A row / B,C column within tile
    int hk = lane >> 4;            // half-wave selector

    // A: 16x64 f16 — lane holds row m; per K-step K = {hk*16..+15}
    const float* ra = x + ((size_t)b * Nn + (size_t)mt * 16 + m) * GH;
    v16h a0, a1;
#pragma unroll
    for (int i = 0; i < 8; ++i) {
        a0[i]     = (_Float16)ra[hk * 16 + i];
        a0[8 + i] = (_Float16)ra[hk * 16 + 8 + i];
        a1[i]     = (_Float16)ra[32 + hk * 16 + i];
        a1[8 + i] = (_Float16)ra[32 + hk * 16 + 8 + i];
    }

    float* outbase = h + ((size_t)b * Nn + (size_t)mt * 16) * GH;
#pragma unroll
    for (int nt = 0; nt < 4; ++nt) {
        // B: 32x16 f16 — lane holds column (nt*16+m), K = k0 + hk*16 .. +15
        // B[k][n] = W[n][k], contiguous along k -> 16 halfs = 32B aligned LDS vector load
        const v16h* wb = (const v16h*)&Wlds[(size_t)(nt * 16 + m) * GH + hk * 16];
        v16h b0 = wb[0];           // k0 = 0
        v16h b1 = wb[2];           // k0 = 32 (32 halfs later)
        v8f c = {};
        c = __builtin_amdgcn_wmma_f32_16x16x32_f16(false, a0, false, b0, (short)0, c,
                                                   false, false);
        c = __builtin_amdgcn_wmma_f32_16x16x32_f16(false, a1, false, b1, (short)0, c,
                                                   false, false);
        // C/D: VGPR r holds row (r + 8*hk), column (nt*16 + m)
        float* out = outbase + nt * 16 + m;
#pragma unroll
        for (int r = 0; r < 8; ++r)
            out[(size_t)(r + 8 * hk) * GH] = c[r];
    }
}

// ---------------- per-layer scratch init ----------------
__global__ void k_edge_init(float* __restrict__ amax, float* __restrict__ denom,
                            float* __restrict__ agg) {
    int idx = blockIdx.x * blockDim.x + threadIdx.x;          // over B*Nn*GH
    if (idx < B * Nn * Hh) { amax[idx] = -1e30f; denom[idx] = 0.0f; }
    agg[idx] = 0.0f;
}

// ---------------- pass 1: per-edge attention logits + scatter-max ----------------
__global__ void k_edge_attn(const float* __restrict__ h, const int* __restrict__ eidx,
                            const int* __restrict__ etype,
                            const float* __restrict__ a_src, const float* __restrict__ a_dst,
                            const float* __restrict__ a_edge, const float* __restrict__ eemb,
                            int layer, float* __restrict__ attn, float* __restrict__ amax) {
    int idx = blockIdx.x * blockDim.x + threadIdx.x;          // over B*Ee
    int b = idx >> 16, e = idx & (Ee - 1);
    int src = eidx[(size_t)(b * 2) * Ee + e];
    int dst = eidx[(size_t)(b * 2 + 1) * Ee + e];
    int et  = etype[(size_t)b * Ee + e];
    const float* hs = h + ((size_t)b * Nn + src) * GH;
    const float* hd = h + ((size_t)b * Nn + dst) * GH;
    __builtin_prefetch(hs, 0, 3);
    __builtin_prefetch(hd, 0, 3);
    const float* As = a_src  + (size_t)layer * Hh * DHd;
    const float* Ad = a_dst  + (size_t)layer * Hh * DHd;
    const float* Ae = a_edge + (size_t)layer * Hh * EEMB;
    const float* Em = eemb   + ((size_t)layer * 3 + et) * EEMB;
#pragma unroll
    for (int hh = 0; hh < Hh; ++hh) {
        float s = 0.0f;
#pragma unroll
        for (int d = 0; d < DHd; ++d)
            s += hs[hh * DHd + d] * As[hh * DHd + d] + hd[hh * DHd + d] * Ad[hh * DHd + d];
#pragma unroll
        for (int d = 0; d < EEMB; ++d) s += Em[d] * Ae[hh * EEMB + d];
        s = s >= 0.0f ? s : 0.2f * s;                          // leaky_relu(0.2)
        attn[(size_t)idx * Hh + hh] = s;
        atomicMaxF(&amax[((size_t)b * Nn + dst) * Hh + hh], s);
    }
}

// ---------------- pass 2: exp + scatter-sum of denom and weighted h_src ----------------
__global__ void k_edge_agg(const float* __restrict__ h, const int* __restrict__ eidx,
                           const float* __restrict__ attn, const float* __restrict__ amax,
                           float* __restrict__ denom, float* __restrict__ agg) {
    int idx = blockIdx.x * blockDim.x + threadIdx.x;          // over B*Ee
    int b = idx >> 16, e = idx & (Ee - 1);
    int src = eidx[(size_t)(b * 2) * Ee + e];
    int dst = eidx[(size_t)(b * 2 + 1) * Ee + e];
    const float* hs = h + ((size_t)b * Nn + src) * GH;
    __builtin_prefetch(hs, 0, 3);
    size_t nd = (size_t)b * Nn + dst;
#pragma unroll
    for (int hh = 0; hh < Hh; ++hh) {
        float m  = fmaxf(amax[nd * Hh + hh], -1e9f);
        float ex = expf(attn[(size_t)idx * Hh + hh] - m);
        atomicAdd(&denom[nd * Hh + hh], ex);
#pragma unroll
        for (int d = 0; d < DHd; ++d)
            atomicAdd(&agg[nd * GH + hh * DHd + d], hs[hh * DHd + d] * ex);
    }
}

// ---------------- finalize: x = gelu(LN(agg/denom + h)), one wave per node ----------------
__global__ void k_finalize(const float* __restrict__ h, const float* __restrict__ agg,
                           const float* __restrict__ denom,
                           const float* __restrict__ g, const float* __restrict__ bt,
                           int layer, float* __restrict__ x) {
    size_t node = (size_t)blockIdx.x * (blockDim.x >> 5) + (threadIdx.x >> 5); // b*Nn+n
    int lane = threadIdx.x & 31;
    int gh0 = lane, gh1 = lane + 32;
    float d0 = denom[node * Hh + (gh0 >> 4)] + 1e-10f;
    float d1 = denom[node * Hh + (gh1 >> 4)] + 1e-10f;
    float v0 = agg[node * GH + gh0] / d0 + h[node * GH + gh0];
    float v1 = agg[node * GH + gh1] / d1 + h[node * GH + gh1];
    float s = v0 + v1;
#pragma unroll
    for (int o = 16; o >= 1; o >>= 1) s += __shfl_xor(s, o, 32);
    float mean = s * (1.0f / 64.0f);
    float q0 = v0 - mean, q1 = v1 - mean;
    float qs = q0 * q0 + q1 * q1;
#pragma unroll
    for (int o = 16; o >= 1; o >>= 1) qs += __shfl_xor(qs, o, 32);
    float inv = rsqrtf(qs * (1.0f / 64.0f) + 1e-5f);
    const float* G  = g  + layer * GH;
    const float* Bt = bt + layer * GH;
    x[node * GH + gh0] = gelu_exact(q0 * inv * G[gh0] + Bt[gh0]);
    x[node * GH + gh1] = gelu_exact(q1 * inv * G[gh1] + Bt[gh1]);
}

// ---------------- head: gat_out for center node + build fused vector ----------------
__global__ void k_fuse(const float* __restrict__ x, const float* __restrict__ rnafm,
                       const float* __restrict__ edit, const float* __restrict__ hand,
                       const float* __restrict__ w, const float* __restrict__ bias,
                       float* __restrict__ fused) {
    int b = blockIdx.x, t = threadIdx.x;
    const float* emb = x + ((size_t)b * Nn + Nn / 2) * GH;    // center node embedding
    float* f = fused + (size_t)b * DF;
    if (t < 64) {
        float s = bias[t];
#pragma unroll
        for (int k = 0; k < GH; ++k) s += emb[k] * w[t * GH + k];
        f[640 + t] = s;
    }
    for (int i = t; i < 640; i += 256) { f[i] = rnafm[b * 640 + i]; f[704 + i] = edit[b * 640 + i]; }
    if (t < 40) f[1344 + t] = hand[b * 40 + t];
}

// ---------------- h1 = LN(gelu(fused @ enc1ᵀ + b)), one block per batch row ----------------
__global__ void k_enc1(const float* __restrict__ fused, const float* __restrict__ w,
                       const float* __restrict__ bias, const float* __restrict__ g,
                       const float* __restrict__ bt, float* __restrict__ h1) {
    __shared__ float tmp[256];
    int b = blockIdx.x, t = threadIdx.x;
    const float* f  = fused + (size_t)b * DF;
    const float* wr = w + (size_t)t * DF;
    float s = bias[t];
    for (int k = 0; k < DF; ++k) s += f[k] * wr[k];
    s = gelu_exact(s);
    tmp[t] = s; __syncthreads();
    for (int o = 128; o >= 1; o >>= 1) { if (t < o) tmp[t] += tmp[t + o]; __syncthreads(); }
    float mean = tmp[0] * (1.0f / 256.0f); __syncthreads();
    float q = s - mean;
    tmp[t] = q * q; __syncthreads();
    for (int o = 128; o >= 1; o >>= 1) { if (t < o) tmp[t] += tmp[t + o]; __syncthreads(); }
    float inv = rsqrtf(tmp[0] * (1.0f / 256.0f) + 1e-5f);
    h1[(size_t)b * 256 + t] = q * inv * g[t] + bt[t];
}

// ---------------- shared + binary + adapters + classifier heads ----------------
__global__ void k_heads(const float* __restrict__ h1,
                        const float* __restrict__ e2w, const float* __restrict__ e2b,
                        const float* __restrict__ binw, const float* __restrict__ binb,
                        const float* __restrict__ a1w, const float* __restrict__ a1b,
                        const float* __restrict__ a2w, const float* __restrict__ a2b,
                        const float* __restrict__ c1w, const float* __restrict__ c1b,
                        const float* __restrict__ c2w, const float* __restrict__ c2b,
                        float* __restrict__ out) {
    __shared__ float sh[128];
    __shared__ float a1[5 * 32];
    __shared__ float c1[64];
    int b = blockIdx.x, t = threadIdx.x;
    const float* hr = h1 + (size_t)b * 256;
    if (t < 128) {
        float s = e2b[t];
        const float* wr = e2w + (size_t)t * 256;
        for (int k = 0; k < 256; ++k) s += hr[k] * wr[k];
        s = gelu_exact(s);
        sh[t] = s;
        out[96 + (size_t)b * 128 + t] = s;                    // 'shared' output
    }
    __syncthreads();
    if (t < 160) {                                            // adapter layer 1: [5][32]
        int hh = t >> 5, i = t & 31;
        float s = a1b[hh * 32 + i];
        const float* wr = a1w + ((size_t)hh * 32 + i) * 128;
        for (int k = 0; k < 128; ++k) s += sh[k] * wr[k];
        a1[t] = gelu_exact(s);
    } else if (t < 224) {                                     // classifier hidden: [64]
        int o = t - 160;
        float s = c1b[o];
        const float* wr = c1w + (size_t)o * 128;
        for (int k = 0; k < 128; ++k) s += sh[k] * wr[k];
        c1[o] = gelu_exact(s);
    }
    __syncthreads();
    if (t == 0) {
        float s = binb[0];
        for (int k = 0; k < 128; ++k) s += sh[k] * binw[k];
        out[b] = s;                                           // binary
    }
    if (t >= 32 && t < 37) {
        int hh = t - 32;
        float s = a2b[hh];
        for (int i = 0; i < 32; ++i) s += a1[hh * 32 + i] * a2w[hh * 32 + i];
        out[8 + b * 5 + hh] = s;                              // per_enz
    }
    if (t >= 64 && t < 70) {
        int o = t - 64;
        float s = c2b[o];
        for (int k = 0; k < 64; ++k) s += c1[k] * c2w[o * 64 + k];
        out[48 + b * 6 + o] = s;                              // cls
    }
}

extern "C" void kernel_launch(void* const* d_in, const int* in_sizes, int n_in,
                              void* d_out, int out_size, void* d_ws, size_t ws_size,
                              hipStream_t stream) {
    (void)in_sizes; (void)n_in; (void)out_size; (void)ws_size;
    const float* rnafm = (const float*)d_in[0];
    const float* edit  = (const float*)d_in[1];
    const float* hand  = (const float*)d_in[2];
    const float* nf    = (const float*)d_in[3];
    const int*   eidx  = (const int*)d_in[4];
    const int*   etyp  = (const int*)d_in[5];
    const float* npw   = (const float*)d_in[6];
    const float* npb   = (const float*)d_in[7];
    const float* gatW  = (const float*)d_in[8];
    const float* a_src = (const float*)d_in[9];
    const float* a_dst = (const float*)d_in[10];
    const float* a_edg = (const float*)d_in[11];
    const float* eemb  = (const float*)d_in[12];
    const float* ln_g  = (const float*)d_in[13];
    const float* ln_b  = (const float*)d_in[14];
    const float* goutw = (const float*)d_in[15];
    const float* goutb = (const float*)d_in[16];
    const float* e1w   = (const float*)d_in[17];
    const float* e1b   = (const float*)d_in[18];
    const float* elng  = (const float*)d_in[19];
    const float* elnb  = (const float*)d_in[20];
    const float* e2w   = (const float*)d_in[21];
    const float* e2b   = (const float*)d_in[22];
    const float* binw  = (const float*)d_in[23];
    const float* binb  = (const float*)d_in[24];
    const float* a1w   = (const float*)d_in[25];
    const float* a1b   = (const float*)d_in[26];
    const float* a2w   = (const float*)d_in[27];
    const float* a2b   = (const float*)d_in[28];
    const float* c1w   = (const float*)d_in[29];
    const float* c1b   = (const float*)d_in[30];
    const float* c2w   = (const float*)d_in[31];
    const float* c2b   = (const float*)d_in[32];
    float* out = (float*)d_out;

    char* ws = (char*)d_ws;
    size_t off = 0;
    auto take = [&](size_t bytes) {
        void* p = ws + off;
        off = (off + bytes + 255) & ~(size_t)255;
        return p;
    };
    float* x     = (float*)take((size_t)B * Nn * GH * 4);   // 8 MB
    float* hbuf  = (float*)take((size_t)B * Nn * GH * 4);   // 8 MB
    float* agg   = (float*)take((size_t)B * Nn * GH * 4);   // 8 MB
    float* attn  = (float*)take((size_t)B * Ee * Hh * 4);   // 8 MB
    float* amax  = (float*)take((size_t)B * Nn * Hh * 4);   // 512 KB
    float* denom = (float*)take((size_t)B * Nn * Hh * 4);   // 512 KB
    float* fused = (float*)take((size_t)B * DF * 4);
    float* h1    = (float*)take((size_t)B * 256 * 4);

    k_node_proj<<<(B * Nn * GH) / 256, 256, 0, stream>>>(nf, npw, npb, x);
    for (int l = 0; l < 3; ++l) {
        k_gemm_h  <<<(B * (Nn / 16)) / 8, 256, 0, stream>>>(x, gatW, l, hbuf);
        k_edge_init<<<(B * Nn * GH) / 256, 256, 0, stream>>>(amax, denom, agg);
        k_edge_attn<<<(B * Ee) / 256, 256, 0, stream>>>(hbuf, eidx, etyp, a_src, a_dst,
                                                        a_edg, eemb, l, attn, amax);
        k_edge_agg<<<(B * Ee) / 256, 256, 0, stream>>>(hbuf, eidx, attn, amax, denom, agg);
        k_finalize<<<(B * Nn) / 8, 256, 0, stream>>>(hbuf, agg, denom, ln_g, ln_b, l, x);
    }
    k_fuse <<<B, 256, 0, stream>>>(x, rnafm, edit, hand, goutw, goutb, fused);
    k_enc1 <<<B, 256, 0, stream>>>(fused, e1w, e1b, elng, elnb, h1);
    k_heads<<<B, 256, 0, stream>>>(h1, e2w, e2b, binw, binb, a1w, a1b, a2w, a2b,
                                   c1w, c1b, c2w, c2b, out);
}
